// GAT_42786464203341
// MI455X (gfx1250) — matrix operand order repaired and checked by reading, not compiled, not run on hardware
//
#include <hip/hip_runtime.h>
#include <hip/hip_bf16.h>
#include <math.h>

constexpr int kB = 2, kT = 265, kNOBJ = 9;
constexpr int kD = 1536, kH = 16, kDH = 96;
constexpr int kV = 8192, kL = 3, kFF = 6144;
constexpr int kN = kB * kT;   // 530 nodes

typedef __attribute__((ext_vector_type(16))) __bf16 bf16x16;
typedef __attribute__((ext_vector_type(8)))  float  f32x8;

union Frag { unsigned int u[8]; bf16x16 v; };

__device__ __forceinline__ unsigned short f2bf(float f) {
  unsigned int u = __float_as_uint(f);
  u += 0x7fffu + ((u >> 16) & 1u);          // round-to-nearest-even
  return (unsigned short)(u >> 16);
}

// ---------------------------------------------------------------------------
// Embedding assembly: x[n, :] = concat(obj/pos embeds | tok embeds) + pos_emb
// ---------------------------------------------------------------------------
__global__ __launch_bounds__(256)
void embed_k(const int* __restrict__ zidx, const int* __restrict__ objs,
             const int* __restrict__ poss, const float* __restrict__ tok,
             const float* __restrict__ objw, const float* __restrict__ posw,
             const float* __restrict__ pemb, float* __restrict__ x) {
  const int n = blockIdx.x;
  const int b = n / kT, t = n - b * kT;
  for (int c = threadIdx.x; c < kD; c += 256) {
    float v;
    if (t < kNOBJ) {
      if (c < 1024)       v = objw[(size_t)objs[b * kNOBJ + t] * 1024 + c];
      else if (c < 1280)  v = posw[(size_t)poss[b * 2 * kNOBJ + t] * 256 + (c - 1024)];
      else                v = posw[(size_t)poss[b * 2 * kNOBJ + kNOBJ + t] * 256 + (c - 1280)];
    } else {
      v = tok[(size_t)zidx[b * (kT - kNOBJ) + (t - kNOBJ)] * kD + c];
    }
    x[(size_t)n * kD + c] = v + pemb[(size_t)t * kD + c];
  }
}

// ---------------------------------------------------------------------------
// LayerNorm (fp32 in) -> bf16 out
// ---------------------------------------------------------------------------
__global__ __launch_bounds__(256)
void ln_bf16(const float* __restrict__ X, const float* __restrict__ g,
             const float* __restrict__ b, unsigned short* __restrict__ Y, int Dn) {
  __shared__ float red[256];
  const int r = blockIdx.x, tid = threadIdx.x;
  const float* xr = X + (size_t)r * Dn;
  float s = 0.f, s2 = 0.f;
  for (int c = tid; c < Dn; c += 256) { float v = xr[c]; s += v; s2 += v * v; }
  red[tid] = s; __syncthreads();
  for (int st = 128; st > 0; st >>= 1) { if (tid < st) red[tid] += red[tid + st]; __syncthreads(); }
  const float mean = red[0] / Dn; __syncthreads();
  red[tid] = s2; __syncthreads();
  for (int st = 128; st > 0; st >>= 1) { if (tid < st) red[tid] += red[tid + st]; __syncthreads(); }
  const float var = red[0] / Dn - mean * mean;
  const float rs = rsqrtf(var + 1e-5f);
  for (int c = tid; c < Dn; c += 256)
    Y[(size_t)r * Dn + c] = f2bf((xr[c] - mean) * rs * g[c] + b[c]);
}

// ---------------------------------------------------------------------------
// bf16 WMMA GEMM: C[N,M] = A_bf16[N,K] @ W_f32->bf16[K,M]   (weight-streaming)
// MODE 0: Cf = acc          MODE 1: Cf += acc + bias        MODE 2: Cb = bf16(gelu(acc+bias))
// Block tile 128x128, K-tile 32, 8 waves of 32 (2x4), each wave 4x2 WMMA tiles.
// Double-buffered LDS; global loads are stashed to LDS immediately so the
// staging registers have a minimal live range (no scratch spills).
// ---------------------------------------------------------------------------
constexpr int BM = 128, BN = 128, BK = 32, LDAS = 40, LDBS = 40;

template <int MODE>
__global__ __launch_bounds__(256)
void gemm_bf16(const unsigned short* __restrict__ A, const float* __restrict__ W,
               const float* __restrict__ bias, float* __restrict__ Cf,
               unsigned short* __restrict__ Cb, int Nrows, int K, int M) {
  __shared__ unsigned short As[2][BM * LDAS];
  __shared__ unsigned short Bs[2][BN * LDBS];

  const int tid  = threadIdx.x;
  const int row0 = blockIdx.y * BM;
  const int col0 = blockIdx.x * BN;
  const int wid  = tid >> 5, lane = tid & 31;
  const int grp  = lane >> 4, l16 = lane & 15;
  const int rbase = (wid >> 2) * 64;      // wave M offset (2 waves along M)
  const int nbase = (wid & 3) * 32;       // wave N offset (4 waves along N)

  f32x8 acc[4][2] = {};
  const int KT = K / BK;

  // ---- tile fetch + immediate LDS stash (short register live range) ----
  auto fetch_tile = [&](int kt, int buf) {
#pragma unroll
    for (int p = 0; p < 2; ++p) {
      const int c = tid + p * 256, r = c >> 2, kc = (c & 3) * 8;
      const int gr = row0 + r;
      uint4 a; a.x = a.y = a.z = a.w = 0u;
      if (gr < Nrows) a = *(const uint4*)(A + (size_t)gr * K + kt * BK + kc);
      *(uint4*)&As[buf][r * LDAS + kc] = a;
    }
#pragma unroll
    for (int p = 0; p < 4; ++p) {
      const int c = tid + p * 256, k = c >> 5, nc = (c & 31) * 4;
      const float4 w = *(const float4*)(W + (size_t)(kt * BK + k) * M + col0 + nc);
      Bs[buf][(nc + 0) * LDBS + k] = f2bf(w.x);
      Bs[buf][(nc + 1) * LDBS + k] = f2bf(w.y);
      Bs[buf][(nc + 2) * LDBS + k] = f2bf(w.z);
      Bs[buf][(nc + 3) * LDBS + k] = f2bf(w.w);
    }
  };

  fetch_tile(0, 0);
  __syncthreads();

  for (int kt = 0; kt < KT; ++kt) {
    const int cur = kt & 1;
    if (kt + 1 < KT) {
      if (kt + 2 < KT)  // stream weights ahead of demand (global_prefetch_b8)
        __builtin_prefetch(W + (size_t)((kt + 2) * BK + (tid >> 3)) * M + col0 + (tid & 7) * 16, 0, 1);
      fetch_tile(kt + 1, cur ^ 1);
    }

    // ---- compute on buffer `cur` ----
    Frag bF[2];
#pragma unroll
    for (int ni = 0; ni < 2; ++ni) {  // B 32x16: lane=col N, halves = K(grp*16 .. +15)
      const uint4* pb = (const uint4*)&Bs[cur][(nbase + ni * 16 + l16) * LDBS + grp * 16];
      const uint4 b0 = pb[0], b1 = pb[1];
      bF[ni].u[0] = b0.x; bF[ni].u[1] = b0.y; bF[ni].u[2] = b0.z; bF[ni].u[3] = b0.w;
      bF[ni].u[4] = b1.x; bF[ni].u[5] = b1.y; bF[ni].u[6] = b1.z; bF[ni].u[7] = b1.w;
    }
#pragma unroll
    for (int mi = 0; mi < 4; ++mi) {  // A 16x32: lane=row M, halves = K(grp*16 .. +15)
      Frag aF;
      const int ar = (rbase + mi * 16 + l16) * LDAS;
      const uint4 a0 = *(const uint4*)&As[cur][ar + grp * 16];
      const uint4 a1 = *(const uint4*)&As[cur][ar + grp * 16 + 8];
      aF.u[0] = a0.x; aF.u[1] = a0.y; aF.u[2] = a0.z; aF.u[3] = a0.w;
      aF.u[4] = a1.x; aF.u[5] = a1.y; aF.u[6] = a1.z; aF.u[7] = a1.w;
#pragma unroll
      for (int ni = 0; ni < 2; ++ni)
        acc[mi][ni] = __builtin_amdgcn_wmma_f32_16x16x32_bf16(
            false, aF.v, false, bF[ni].v, (short)0, acc[mi][ni], false, false);
    }
    __syncthreads();
  }

  // ---- epilogue: C/D layout: VGPR v -> row v+grp*8, col l16 ----
#pragma unroll
  for (int mi = 0; mi < 4; ++mi) {
#pragma unroll
    for (int ni = 0; ni < 2; ++ni) {
      const int gc = col0 + nbase + ni * 16 + l16;
      const float bv = (MODE != 0) ? bias[gc] : 0.0f;
#pragma unroll
      for (int v = 0; v < 8; ++v) {
        const int gr = row0 + rbase + mi * 16 + grp * 8 + v;
        if (gr >= Nrows) continue;
        const float val = acc[mi][ni][v];
        const size_t idx = (size_t)gr * M + gc;
        if (MODE == 0) {
          Cf[idx] = val;
        } else if (MODE == 1) {
          Cf[idx] += val + bv;                       // residual += proj + bias
        } else {
          float t = val + bv;                        // exact GELU
          t = 0.5f * t * (1.0f + erff(t * 0.70710678118654752f));
          Cb[idx] = f2bf(t);
        }
      }
    }
  }
}

// ---------------------------------------------------------------------------
// el/er: per-head dot products of Wh with a_l / a_r
// ---------------------------------------------------------------------------
__global__ __launch_bounds__(256)
void elr_k(const float* __restrict__ Wh, const float* __restrict__ al,
           const float* __restrict__ ar, float* __restrict__ el,
           float* __restrict__ er) {
  const int n = blockIdx.x, tid = threadIdx.x;
  const int h = tid >> 4, sl = tid & 15;
  float sL = 0.f, sR = 0.f;
  const float* wr = Wh + (size_t)n * kD + h * kDH;
  for (int d = sl; d < kDH; d += 16) {
    const float w = wr[d];
    sL += w * al[h * kDH + d];
    sR += w * ar[h * kDH + d];
  }
  for (int m = 8; m >= 1; m >>= 1) {
    sL += __shfl_xor(sL, m, 16);
    sR += __shfl_xor(sR, m, 16);
  }
  if (sl == 0) { el[(size_t)n * kH + h] = sL; er[(size_t)n * kH + h] = sR; }
}

// ---------------------------------------------------------------------------
// Dense block-diagonal GAT attention: one block per (b, h, dst i)
// ---------------------------------------------------------------------------
__global__ __launch_bounds__(128)
void attn_k(const float* __restrict__ el, const float* __restrict__ er,
            const float* __restrict__ Wh, unsigned short* __restrict__ agg) {
  __shared__ float sc[kT];
  __shared__ float red[128];
  const int i = blockIdx.x, h = blockIdx.y, b = blockIdx.z;
  const int tid = threadIdx.x;
  const int ni = b * kT + i;
  const float eri = er[(size_t)ni * kH + h];

  float lm = -3.4e38f;
  for (int j = tid; j < kT; j += 128) {
    float e = el[(size_t)(b * kT + j) * kH + h] + eri;
    e = e > 0.f ? e : 0.2f * e;                     // leaky_relu(0.2)
    sc[j] = e;
    lm = fmaxf(lm, e);
  }
  red[tid] = lm; __syncthreads();
  for (int st = 64; st > 0; st >>= 1) { if (tid < st) red[tid] = fmaxf(red[tid], red[tid + st]); __syncthreads(); }
  const float m = red[0]; __syncthreads();

  float ls = 0.f;
  for (int j = tid; j < kT; j += 128) { const float ex = __expf(sc[j] - m); sc[j] = ex; ls += ex; }
  red[tid] = ls; __syncthreads();
  for (int st = 64; st > 0; st >>= 1) { if (tid < st) red[tid] += red[tid + st]; __syncthreads(); }
  const float inv = 1.f / red[0];
  __syncthreads();

  if (tid < kDH) {
    float acc = 0.f;
    const float* wp = Wh + (size_t)(b * kT) * kD + h * kDH + tid;
    for (int j = 0; j < kT; ++j) acc += sc[j] * wp[(size_t)j * kD];
    agg[(size_t)ni * kD + h * kDH + tid] = f2bf(acc * inv);
  }
}

// ---------------------------------------------------------------------------
extern "C" void kernel_launch(void* const* d_in, const int* in_sizes, int n_in,
                              void* d_out, int out_size, void* d_ws, size_t ws_size,
                              hipStream_t stream) {
  const int*   zidx   = (const int*)d_in[0];
  const int*   objs   = (const int*)d_in[1];
  const int*   poss   = (const int*)d_in[2];
  // d_in[3]/d_in[4] (src/dst) unused: graph is dense block-diagonal by construction
  const float* tok    = (const float*)d_in[5];
  const float* objw   = (const float*)d_in[6];
  const float* posw   = (const float*)d_in[7];
  const float* pemb   = (const float*)d_in[8];
  const float* W_attn = (const float*)d_in[9];
  const float* a_l    = (const float*)d_in[10];
  const float* a_r    = (const float*)d_in[11];
  const float* W_proj = (const float*)d_in[12];
  const float* b_proj = (const float*)d_in[13];
  const float* ln1_g  = (const float*)d_in[14];
  const float* ln1_b  = (const float*)d_in[15];
  const float* ln2_g  = (const float*)d_in[16];
  const float* ln2_b  = (const float*)d_in[17];
  const float* W_fc   = (const float*)d_in[18];
  const float* b_fc   = (const float*)d_in[19];
  const float* W_out  = (const float*)d_in[20];
  const float* b_out  = (const float*)d_in[21];
  const float* lnf_g  = (const float*)d_in[22];
  const float* lnf_b  = (const float*)d_in[23];
  const float* head_w = (const float*)d_in[24];

  char* ws = (char*)d_ws;
  size_t off = 0;
  auto take = [&](size_t bytes) -> void* {
    void* p = ws + off;
    off = (off + bytes + 255) & ~(size_t)255;
    return p;
  };
  float*          x    = (float*)take((size_t)kN * kD * 4);
  unsigned short* hbuf = (unsigned short*)take((size_t)kN * kD * 2);
  float*          Wh   = (float*)take((size_t)kN * kD * 4);
  float*          el   = (float*)take((size_t)kN * kH * 4);
  float*          er   = (float*)take((size_t)kN * kH * 4);
  unsigned short* aggb = (unsigned short*)take((size_t)kN * kD * 2);
  unsigned short* tb   = (unsigned short*)take((size_t)kN * kFF * 2);

  const int GR = (kN + BM - 1) / BM;  // 5 row blocks

  embed_k<<<kN, 256, 0, stream>>>(zidx, objs, poss, tok, objw, posw, pemb, x);

  for (int l = 0; l < kL; ++l) {
    ln_bf16<<<kN, 256, 0, stream>>>(x, ln1_g + l * kD, ln1_b + l * kD, hbuf, kD);
    gemm_bf16<0><<<dim3(kD / BN, GR), 256, 0, stream>>>(
        hbuf, W_attn + (size_t)l * kD * kD, nullptr, Wh, nullptr, kN, kD, kD);
    elr_k<<<kN, 256, 0, stream>>>(Wh, a_l + (size_t)l * kH * kDH,
                                  a_r + (size_t)l * kH * kDH, el, er);
    attn_k<<<dim3(kT, kH, kB), 128, 0, stream>>>(el, er, Wh, aggb);
    gemm_bf16<1><<<dim3(kD / BN, GR), 256, 0, stream>>>(
        aggb, W_proj + (size_t)l * kD * kD, b_proj + l * kD, x, nullptr, kN, kD, kD);
    ln_bf16<<<kN, 256, 0, stream>>>(x, ln2_g + l * kD, ln2_b + l * kD, hbuf, kD);
    gemm_bf16<2><<<dim3(kFF / BN, GR), 256, 0, stream>>>(
        hbuf, W_fc + (size_t)l * kD * kFF, b_fc + l * kFF, nullptr, tb, kN, kD, kFF);
    gemm_bf16<1><<<dim3(kD / BN, GR), 256, 0, stream>>>(
        tb, W_out + (size_t)l * kFF * kD, b_out + l * kD, x, nullptr, kN, kFF, kD);
  }

  ln_bf16<<<kN, 256, 0, stream>>>(x, lnf_g, lnf_b, hbuf, kD);
  gemm_bf16<0><<<dim3(kV / BN, GR), 256, 0, stream>>>(
      hbuf, head_w, nullptr, (float*)d_out, nullptr, kN, kD, kV);
}
